// MirrorAttention_1769526526124
// MI455X (gfx1250) — matrix operand order repaired and checked by reading, not compiled, not run on hardware
//
#include <hip/hip_runtime.h>
#include <hip/hip_bf16.h>
#include <stdint.h>

typedef __bf16 bf16_t;
typedef __attribute__((ext_vector_type(16))) __bf16 v16bf;
typedef __attribute__((ext_vector_type(8)))  __bf16 v8bf;
typedef __attribute__((ext_vector_type(8)))  float  v8f;

constexpr int kB   = 8;
constexpr int kC   = 512;
constexpr int kMid = 128;
constexpr int kN   = 48 * 48;  // 2304
constexpr float kEps = 1e-5f;

// ---------------------------------------------------------------------------
// CDNA5 WMMA fragment loaders (wave32), fully vectorized:
//  A frag (16Mx32K bf16) from row-major [R][K]: lane&15 = row,
//    one contiguous 16-element (32B) chunk at K offset (lane&16)?16:0.
//  B frag (32Kx16N bf16) from transposed storage [Col][K]: lane&15 = col,
//    one contiguous 16-element (32B) chunk at K offset (lane&16)?16:0.
//  C/D frag (16x16 f32): lane&15 = col, VGPR r = row r + ((lane&16)?8:0).
// ---------------------------------------------------------------------------
__device__ __forceinline__ v16bf load_a_frag(const bf16_t* __restrict__ src, int ld) {
  const int lane = threadIdx.x & 31;
  return *reinterpret_cast<const v16bf*>(src + (long)(lane & 15) * ld +
                                         ((lane & 16) ? 16 : 0));
}

__device__ __forceinline__ v16bf load_b_frag(const bf16_t* __restrict__ src, int ld) {
  const int lane = threadIdx.x & 31;
  return *reinterpret_cast<const v16bf*>(src + (long)(lane & 15) * ld +
                                         ((lane & 16) ? 16 : 0));
}

#define WMMA_BF16(A, Bm, Cacc) \
  __builtin_amdgcn_wmma_f32_16x16x32_bf16(false, (A), false, (Bm), (short)0, (Cacc), false, false)

// ---------------------------------------------------------------------------
// Kernel W: one-time weight prep. fp32 [MO][CIN] -> bf16, with the eval-BN
// scale gamma/sqrt(1+eps) folded into rows; bias2[m] = bias*scale + beta.
// ---------------------------------------------------------------------------
__global__ __launch_bounds__(256) void wconv_kernel(
    const float* __restrict__ W, const float* __restrict__ bias,
    const float* __restrict__ gamma, const float* __restrict__ beta,
    int MO, int CIN, bf16_t* __restrict__ Wout, float* __restrict__ bias_out) {
  const float bnscale = rsqrtf(1.0f + kEps);
  const long t  = (long)blockIdx.x * 256 + threadIdx.x;
  const long e0 = t * 8;
  if (e0 < (long)MO * CIN) {
    const int row = (int)(e0 / CIN);
    const float s = gamma ? gamma[row] * bnscale : 1.0f;
    const float4 p0 = *reinterpret_cast<const float4*>(W + e0);
    const float4 p1 = *reinterpret_cast<const float4*>(W + e0 + 4);
    v8bf o;
    o[0] = (__bf16)(p0.x * s); o[1] = (__bf16)(p0.y * s);
    o[2] = (__bf16)(p0.z * s); o[3] = (__bf16)(p0.w * s);
    o[4] = (__bf16)(p1.x * s); o[5] = (__bf16)(p1.y * s);
    o[6] = (__bf16)(p1.z * s); o[7] = (__bf16)(p1.w * s);
    *reinterpret_cast<v8bf*>(Wout + e0) = o;
  }
  if (t < MO)
    bias_out[t] = gamma ? bias[t] * (gamma[t] * bnscale) + beta[t] : bias[t];
}

// ---------------------------------------------------------------------------
// Kernel 0: x[B][C][N] fp32 -> xT[B][N][C] bf16, 64x64 tiles staged through
// LDS with CDNA5 async global->LDS copies (ASYNCcnt), transposed read-out.
// ---------------------------------------------------------------------------
__global__ __launch_bounds__(256) void transpose_kernel(
    const float* __restrict__ X, bf16_t* __restrict__ XT) {
  __shared__ float tile[64][65];  // +1 pad: conflict-free transposed reads
  const int p0 = blockIdx.x * 64;
  const int c0 = blockIdx.y * 64;
  const int b  = blockIdx.z;
  const int t  = threadIdx.x;

#pragma unroll
  for (int q = 0; q < 4; ++q) {
    const int u   = t + q * 256;
    const int cr  = u >> 4;
    const int seg = u & 15;
    const float* gsrc = X + ((long)b * kC + c0 + cr) * kN + p0 + seg * 4;
    const uint32_t lds_off = (uint32_t)(uintptr_t)(&tile[cr][seg * 4]);
    asm volatile("global_load_async_to_lds_b128 %0, %1, off"
                 :: "v"(lds_off), "v"((uint64_t)(uintptr_t)gsrc)
                 : "memory");
  }
  asm volatile("s_wait_asynccnt 0" ::: "memory");
  __syncthreads();

  const int p  = t >> 2;
  const int cs = (t & 3) * 16;
  v16bf o;
#pragma unroll
  for (int e = 0; e < 16; ++e) o[e] = (__bf16)tile[cs + e][p];
  *reinterpret_cast<v16bf*>(XT + ((long)b * kN + p0 + p) * kC + c0 + cs) = o;
}

// ---------------------------------------------------------------------------
// Kernel 1: projection  out[m,p] = sum_c W'[m,c]*x[c,p] + bias2[m] (+ReLU)
// Two m-tiles per wave share one B fragment; two independent WMMA chains.
// ---------------------------------------------------------------------------
__global__ __launch_bounds__(128) void proj_kernel(
    const bf16_t* __restrict__ XT,    // [B][N][C]
    const bf16_t* __restrict__ Wt,    // [MO][C] bf16, BN-folded
    const float* __restrict__ bias2,  // [MO]
    bf16_t* __restrict__ out, int MO, int transpose_out, int relu) {
  const int wv   = threadIdx.x >> 5;
  const int lane = threadIdx.x & 31;
  const int p0 = (blockIdx.x * 4 + wv) * 16;
  const int m0 = blockIdx.y * 32;
  const int b  = blockIdx.z;
  const bf16_t* xb = XT + ((long)b * kN + p0) * kC;
  const bf16_t* w0 = Wt + (long)m0 * kC;
  const bf16_t* w1 = w0 + 16 * kC;

  v8f acc0 = {}, acc1 = {};
#pragma unroll 4
  for (int kc = 0; kc < kC; kc += 32) {
    v16bf bm = load_b_frag(xb + kc, kC);
    v16bf a0 = load_a_frag(w0 + kc, kC);
    v16bf a1 = load_a_frag(w1 + kc, kC);
    acc0 = WMMA_BF16(a0, bm, acc0);
    acc1 = WMMA_BF16(a1, bm, acc1);
  }

  const int colp = p0 + (lane & 15);
  const int rofs = (lane & 16) ? 8 : 0;
  v8f accs[2] = {acc0, acc1};
#pragma unroll
  for (int h = 0; h < 2; ++h) {
    const int mt = m0 + 16 * h;
    if (transpose_out) {
      v8bf pk;
#pragma unroll
      for (int r = 0; r < 8; ++r) {
        float v = accs[h][r] + bias2[mt + r + rofs];
        if (relu) v = v > 0.0f ? v : 0.0f;
        pk[r] = (__bf16)v;
      }
      *reinterpret_cast<v8bf*>(out + ((long)b * kN + colp) * MO + mt + rofs) = pk;
    } else {
#pragma unroll
      for (int r = 0; r < 8; ++r) {
        const int m = mt + r + rofs;
        float v = accs[h][r] + bias2[m];
        if (relu) v = v > 0.0f ? v : 0.0f;
        out[((long)b * MO + m) * kN + colp] = (__bf16)v;
      }
    }
  }
}

// ---------------------------------------------------------------------------
// Kernel 2: scores + row softmax, writing P^T[j][i].
// Queries (16x128) resident in 32 VGPRs; 2 j-tiles/iter -> two independent
// WMMA chains; one shfl_xor butterfly reduces stats of both tiles at once.
// ---------------------------------------------------------------------------
__global__ __launch_bounds__(128) void attn_kernel(
    const bf16_t* __restrict__ QT,  // [B][N][MID]
    const bf16_t* __restrict__ KT,  // [B][N][MID]
    bf16_t* __restrict__ PT) {      // [B][N_j][N_i]
  const int wv   = threadIdx.x >> 5;
  const int lane = threadIdx.x & 31;
  const int i0 = (blockIdx.x * 4 + wv) * 16;
  const int b  = blockIdx.y;
  const float scale = 0.088388347648318447f;  // 128^-0.5

  v16bf q[4];
#pragma unroll
  for (int kt = 0; kt < 4; ++kt)
    q[kt] = load_a_frag(QT + ((long)b * kN + i0) * kMid + kt * 32, kMid);

  const bf16_t* kbase = KT + (long)b * kN * kMid;

  float rm[8], rs[8];
#pragma unroll
  for (int r = 0; r < 8; ++r) { rm[r] = -3.0e38f; rs[r] = 0.0f; }

  // pass 1: online row stats (max, sumexp) over all j
  for (int j0 = 0; j0 < kN; j0 += 32) {
    const bf16_t* kb0 = kbase + (long)j0 * kMid;
    const bf16_t* kb1 = kb0 + 16 * kMid;
    v8f acc0 = {}, acc1 = {};
#pragma unroll
    for (int kt = 0; kt < 4; ++kt) {
      v16bf b0 = load_b_frag(kb0 + kt * 32, kMid);
      v16bf b1 = load_b_frag(kb1 + kt * 32, kMid);
      acc0 = WMMA_BF16(q[kt], b0, acc0);
      acc1 = WMMA_BF16(q[kt], b1, acc1);
    }
#pragma unroll
    for (int r = 0; r < 8; ++r) {
      const float t0 = acc0[r] * scale;
      const float t1 = acc1[r] * scale;
      float tm = fmaxf(t0, t1);
      tm = fmaxf(tm, __shfl_xor(tm, 1, 32));
      tm = fmaxf(tm, __shfl_xor(tm, 2, 32));
      tm = fmaxf(tm, __shfl_xor(tm, 4, 32));
      tm = fmaxf(tm, __shfl_xor(tm, 8, 32));
      const float nm = fmaxf(rm[r], tm);
      float e = __expf(t0 - nm) + __expf(t1 - nm);
      e += __shfl_xor(e, 1, 32);
      e += __shfl_xor(e, 2, 32);
      e += __shfl_xor(e, 4, 32);
      e += __shfl_xor(e, 8, 32);
      rs[r] = rs[r] * __expf(rm[r] - nm) + e;
      rm[r] = nm;
    }
  }
#pragma unroll
  for (int r = 0; r < 8; ++r) rs[r] = 1.0f / rs[r];

  // pass 2: normalized probabilities -> P^T (both maps: 170MB < 192MB L2)
  const int rofs = (lane & 16) ? 8 : 0;
  const int colj = lane & 15;
  for (int j0 = 0; j0 < kN; j0 += 32) {
    const bf16_t* kb0 = kbase + (long)j0 * kMid;
    const bf16_t* kb1 = kb0 + 16 * kMid;
    v8f acc0 = {}, acc1 = {};
#pragma unroll
    for (int kt = 0; kt < 4; ++kt) {
      v16bf b0 = load_b_frag(kb0 + kt * 32, kMid);
      v16bf b1 = load_b_frag(kb1 + kt * 32, kMid);
      acc0 = WMMA_BF16(q[kt], b0, acc0);
      acc1 = WMMA_BF16(q[kt], b1, acc1);
    }
    v8bf pk0, pk1;
#pragma unroll
    for (int r = 0; r < 8; ++r) {
      pk0[r] = (__bf16)(__expf(acc0[r] * scale - rm[r]) * rs[r]);
      pk1[r] = (__bf16)(__expf(acc1[r] * scale - rm[r]) * rs[r]);
    }
    *reinterpret_cast<v8bf*>(PT + ((long)b * kN + j0 + colj) * kN + i0 + rofs) = pk0;
    *reinterpret_cast<v8bf*>(PT + ((long)b * kN + j0 + 16 + colj) * kN + i0 + rofs) = pk1;
  }
}

// ---------------------------------------------------------------------------
// Kernel 3: value GEMM (K = 2304):  O[m,j] = sum_n G[m,n] P[n,j]
// 2x2 register tile (32x32/wave): 4 independent WMMA chains, ~1 vmem/WMMA,
// 2x operand reuse. Stores O^T[B][N][MID] packed b128.
// ---------------------------------------------------------------------------
__global__ __launch_bounds__(128) void ogemm_kernel(
    const bf16_t* __restrict__ G,   // [B][MID][N]
    const bf16_t* __restrict__ PT,  // [B][N_j][N_n]
    bf16_t* __restrict__ OT) {      // [B][N][MID]
  const int wv   = threadIdx.x >> 5;
  const int lane = threadIdx.x & 31;
  const int j0 = (blockIdx.x * 4 + wv) * 32;
  const int m0 = blockIdx.y * 32;
  const int b  = blockIdx.z;
  const bf16_t* ga0 = G + ((long)b * kMid + m0) * kN;
  const bf16_t* ga1 = ga0 + (long)16 * kN;
  const bf16_t* pb0 = PT + ((long)b * kN + j0) * kN;
  const bf16_t* pb1 = pb0 + (long)16 * kN;
  const bf16_t* pfl = pb0 + (long)(lane & 15) * kN;  // per-lane prefetch stream

  v8f a00 = {}, a01 = {}, a10 = {}, a11 = {};
  for (int kn = 0; kn < kN; kn += 32) {
    if (kn + 128 < kN) __builtin_prefetch(pfl + kn + 128, 0, 3);
    v16bf A0 = load_a_frag(ga0 + kn, kN);
    v16bf A1 = load_a_frag(ga1 + kn, kN);
    v16bf B0 = load_b_frag(pb0 + kn, kN);
    v16bf B1 = load_b_frag(pb1 + kn, kN);
    a00 = WMMA_BF16(A0, B0, a00);
    a01 = WMMA_BF16(A0, B1, a01);
    a10 = WMMA_BF16(A1, B0, a10);
    a11 = WMMA_BF16(A1, B1, a11);
  }

  const int colj = j0 + (lane & 15);
  const int rofs = (lane & 16) ? 8 : 0;
  v8f accs[4] = {a00, a01, a10, a11};
#pragma unroll
  for (int h = 0; h < 4; ++h) {
    const int jj = colj + ((h & 1) ? 16 : 0);
    const int mm = m0 + ((h & 2) ? 16 : 0) + rofs;
    v8bf pk;
#pragma unroll
    for (int r = 0; r < 8; ++r) pk[r] = (__bf16)accs[h][r];
    *reinterpret_cast<v8bf*>(OT + ((long)b * kN + jj) * kMid + mm) = pk;
  }
}

// ---------------------------------------------------------------------------
// Kernel 4: output conv + residual:  out[c,j] = sum_m Wf[c,m] O[m,j] + b + x
// Two c-tiles per wave share one B fragment; bf16 pre-converted weights.
// ---------------------------------------------------------------------------
__global__ __launch_bounds__(128) void final_kernel(
    const bf16_t* __restrict__ Wf,    // [C][MID] bf16
    const float* __restrict__ bias,   // [C]
    const bf16_t* __restrict__ OT,    // [B][N][MID]
    const float* __restrict__ X,      // [B][C][N]
    float* __restrict__ Out) {        // [B][C][N]
  const int wv   = threadIdx.x >> 5;
  const int lane = threadIdx.x & 31;
  const int j0 = (blockIdx.x * 4 + wv) * 16;
  const int c0 = blockIdx.y * 32;
  const int b  = blockIdx.z;
  const bf16_t* ob = OT + ((long)b * kN + j0) * kMid;
  const bf16_t* w0 = Wf + (long)c0 * kMid;
  const bf16_t* w1 = w0 + 16 * kMid;

  v8f acc0 = {}, acc1 = {};
#pragma unroll
  for (int kt = 0; kt < 4; ++kt) {
    v16bf bm = load_b_frag(ob + kt * 32, kMid);
    v16bf a0 = load_a_frag(w0 + kt * 32, kMid);
    v16bf a1 = load_a_frag(w1 + kt * 32, kMid);
    acc0 = WMMA_BF16(a0, bm, acc0);
    acc1 = WMMA_BF16(a1, bm, acc1);
  }

  const int colj = j0 + (lane & 15);
  const int rofs = (lane & 16) ? 8 : 0;
  v8f accs[2] = {acc0, acc1};
#pragma unroll
  for (int h = 0; h < 2; ++h) {
#pragma unroll
    for (int r = 0; r < 8; ++r) {
      const int c = c0 + 16 * h + r + rofs;
      const long idx = ((long)b * kC + c) * kN + colj;
      Out[idx] = accs[h][r] + bias[c] + X[idx];
    }
  }
}

// ---------------------------------------------------------------------------
extern "C" void kernel_launch(void* const* d_in, const int* in_sizes, int n_in,
                              void* d_out, int out_size, void* d_ws, size_t ws_size,
                              hipStream_t stream) {
  const float* x    = (const float*)d_in[0];
  const float* x_h  = (const float*)d_in[1];
  const float* x_v  = (const float*)d_in[2];
  const float* Wa   = (const float*)d_in[3];
  const float* ba   = (const float*)d_in[4];
  const float* ga   = (const float*)d_in[5];
  const float* ta   = (const float*)d_in[6];
  const float* Wv   = (const float*)d_in[7];
  const float* bv   = (const float*)d_in[8];
  const float* gv   = (const float*)d_in[9];
  const float* tv   = (const float*)d_in[10];
  const float* Wgav = (const float*)d_in[11];
  const float* bgav = (const float*)d_in[12];
  const float* Wgah = (const float*)d_in[13];
  const float* bgah = (const float*)d_in[14];
  const float* Wfav = (const float*)d_in[15];
  const float* bfav = (const float*)d_in[16];
  const float* Wfah = (const float*)d_in[17];
  const float* bfah = (const float*)d_in[18];

  char* ws = (char*)d_ws;
  const size_t S1 = (size_t)kB * kMid * kN * sizeof(bf16_t);  // 4.5 MiB
  const size_t SX = (size_t)kB * kN * kC * sizeof(bf16_t);    // 18 MiB
  const size_t SA = (size_t)kB * kN * kN * sizeof(bf16_t);    // 81 MiB
  const size_t SW = (size_t)kMid * kC * sizeof(bf16_t);       // 128 KiB
  bf16_t* KTa = (bf16_t*)(ws);            // keys^T   [B][N][MID]
  bf16_t* QvT = (bf16_t*)(ws + 1 * S1);   // f_v^T    [B][N][MID]
  bf16_t* QhT = (bf16_t*)(ws + 2 * S1);   // f_h^T    [B][N][MID]
  bf16_t* gav = (bf16_t*)(ws + 3 * S1);   // [B][MID][N]
  bf16_t* gah = (bf16_t*)(ws + 4 * S1);
  bf16_t* OvT = (bf16_t*)(ws + 5 * S1);   // [B][N][MID]
  bf16_t* OhT = (bf16_t*)(ws + 6 * S1);
  bf16_t* Pav = (bf16_t*)(ws + 7 * S1);   // P^T [B][N][N]
  bf16_t* Pah = (bf16_t*)(ws + 7 * S1 + SA);
  // xT buffers live inside the P region (dead once projections finish)
  bf16_t* xT  = (bf16_t*)(ws + 7 * S1);
  bf16_t* xvT = (bf16_t*)(ws + 7 * S1 + SX);
  bf16_t* xhT = (bf16_t*)(ws + 7 * S1 + 2 * SX);
  // bf16 weights + folded biases appended past the P region
  char* wreg = ws + 7 * S1 + 2 * SA;
  bf16_t* Wa2   = (bf16_t*)(wreg + 0 * SW);
  bf16_t* Wv2   = (bf16_t*)(wreg + 1 * SW);
  bf16_t* Wgav2 = (bf16_t*)(wreg + 2 * SW);
  bf16_t* Wgah2 = (bf16_t*)(wreg + 3 * SW);
  bf16_t* Wfav2 = (bf16_t*)(wreg + 4 * SW);
  bf16_t* Wfah2 = (bf16_t*)(wreg + 5 * SW);
  float* ba2   = (float*)(wreg + 6 * SW);
  float* bv2   = ba2 + kMid;
  float* bgav2 = bv2 + kMid;
  float* bgah2 = bgav2 + kMid;
  float* bfav2 = bgah2 + kMid;
  float* bfah2 = bfav2 + kC;

  const float* kNull = nullptr;

  // W) weight prep: fp32 -> bf16 with BN folding (tiny, one-time)
  const dim3 gw(kMid * kC / (8 * 256));  // 32 blocks
  hipLaunchKernelGGL(wconv_kernel, gw, dim3(256), 0, stream, Wa,   ba,   ga,    ta,    kMid, kC, Wa2,   ba2);
  hipLaunchKernelGGL(wconv_kernel, gw, dim3(256), 0, stream, Wv,   bv,   gv,    tv,    kMid, kC, Wv2,   bv2);
  hipLaunchKernelGGL(wconv_kernel, gw, dim3(256), 0, stream, Wgav, bgav, kNull, kNull, kMid, kC, Wgav2, bgav2);
  hipLaunchKernelGGL(wconv_kernel, gw, dim3(256), 0, stream, Wgah, bgah, kNull, kNull, kMid, kC, Wgah2, bgah2);
  hipLaunchKernelGGL(wconv_kernel, gw, dim3(256), 0, stream, Wfav, bfav, kNull, kNull, kC, kMid, Wfav2, bfav2);
  hipLaunchKernelGGL(wconv_kernel, gw, dim3(256), 0, stream, Wfah, bfah, kNull, kNull, kC, kMid, Wfah2, bfah2);

  // 0) fp32 -> bf16 transposes via async global->LDS staging
  const dim3 gtr(kN / 64, kC / 64, kB);
  hipLaunchKernelGGL(transpose_kernel, gtr, dim3(256), 0, stream, x,   xT);
  hipLaunchKernelGGL(transpose_kernel, gtr, dim3(256), 0, stream, x_v, xvT);
  hipLaunchKernelGGL(transpose_kernel, gtr, dim3(256), 0, stream, x_h, xhT);

  // 1) projections (GEMM over C=512; BN+ReLU folded)
  const dim3 gproj(kN / 64, kMid / 32, kB);
  hipLaunchKernelGGL(proj_kernel, gproj, dim3(128), 0, stream, xT,  Wa2, ba2, KTa, kMid, 1, 1);
  hipLaunchKernelGGL(proj_kernel, gproj, dim3(128), 0, stream, xvT, Wv2, bv2, QvT, kMid, 1, 1);
  hipLaunchKernelGGL(proj_kernel, gproj, dim3(128), 0, stream, xhT, Wv2, bv2, QhT, kMid, 1, 1);
  hipLaunchKernelGGL(proj_kernel, gproj, dim3(128), 0, stream, xT, Wgav2, bgav2, gav, kMid, 0, 0);
  hipLaunchKernelGGL(proj_kernel, gproj, dim3(128), 0, stream, xT, Wgah2, bgah2, gah, kMid, 0, 0);

  // 2) scores + online softmax -> P^T (xT buffers are dead now; reuse region)
  const dim3 gattn(kN / 64, kB);
  hipLaunchKernelGGL(attn_kernel, gattn, dim3(128), 0, stream, QvT, KTa, Pav);
  hipLaunchKernelGGL(attn_kernel, gattn, dim3(128), 0, stream, QhT, KTa, Pah);

  // 3) value GEMMs (K = 2304), P maps served from L2
  const dim3 gog(kN / 128, kMid / 32, kB);
  hipLaunchKernelGGL(ogemm_kernel, gog, dim3(128), 0, stream, gav, Pav, OvT);
  hipLaunchKernelGGL(ogemm_kernel, gog, dim3(128), 0, stream, gah, Pah, OhT);

  // 4) output convs + residual; d_out = concat(o_h, o_v)
  float* out_h = (float*)d_out;
  float* out_v = out_h + (size_t)kB * kC * kN;
  const dim3 gfin(kN / 64, kC / 32, kB);
  hipLaunchKernelGGL(final_kernel, gfin, dim3(128), 0, stream, Wfah2, bfah, OhT, x, out_h);
  hipLaunchKernelGGL(final_kernel, gfin, dim3(128), 0, stream, Wfav2, bfav, OvT, x, out_v);
}
